// StableLMAttention_25177098289609
// MI455X (gfx1250) — matrix operand order repaired and checked by reading, not compiled, not run on hardware
//
#include <hip/hip_runtime.h>

typedef __attribute__((ext_vector_type(16))) __bf16 v16bf;
typedef __attribute__((ext_vector_type(8)))  float  v8f;
typedef unsigned int uint32x4 __attribute__((ext_vector_type(4)));
typedef int          int32x8  __attribute__((ext_vector_type(8)));
typedef int          int32x4  __attribute__((ext_vector_type(4)));

#define B_    2
#define S_    2048
#define H_    2048
#define NH_   32
#define NKV_  8
#define HD_   64
#define ROT_  16
#define G_    (NH_/NKV_)
#define MROWS (B_*S_)   // 4096

#if defined(__gfx1250__)
#if __has_builtin(__builtin_amdgcn_tensor_load_to_lds) && \
    __has_builtin(__builtin_amdgcn_s_wait_tensorcnt)
#define USE_TDM 1
#endif
#endif
#ifndef USE_TDM
#define USE_TDM 0
#endif

// ---------------------------------------------------------------------------
// f32 -> bf16 elementwise convert
// ---------------------------------------------------------------------------
__global__ void cvt_bf16_kernel(const float* __restrict__ src,
                                __bf16* __restrict__ dst, int n) {
  int i = blockIdx.x * blockDim.x + threadIdx.x;
  if (i < n) dst[i] = (__bf16)src[i];
}

#if USE_TDM
// Issue a TDM 2-D tile load (rows x 32 bf16) global -> LDS.
// D# layout per CDNA5 ISA ch.8: group0 = {flags/count, lds_addr, gaddr, type},
// group1 = {mask/data_size, dims, tile dims, strides}. Groups 2/3 zero (2-D).
// This toolchain's builtin takes 6 args (extra int32x8 group before cpol).
__device__ __forceinline__ void tdm_load_tile_2d(unsigned lds_off,
                                                 const __bf16* gptr,
                                                 unsigned tensor_d0,
                                                 unsigned tensor_d1,
                                                 unsigned stride_elems,
                                                 unsigned tile_d0,
                                                 unsigned tile_d1) {
  unsigned long long ga = (unsigned long long)(size_t)gptr;
  uint32x4 g0;
  g0[0] = 1u;                                            // count=1, user desc
  g0[1] = lds_off;                                       // lds_addr (bytes)
  g0[2] = (unsigned)(ga & 0xffffffffu);                  // global_addr[31:0]
  g0[3] = (unsigned)((ga >> 32) & 0x01ffffffu) | (2u << 30); // [56:32] | type=2
  int32x8 g1;
  g1[0] = (int)(1u << 16);                               // data_size=1 (2B)
  g1[1] = (int)((tensor_d0 & 0xffffu) << 16);            // tensor_dim0[15:0]
  g1[2] = (int)((tensor_d0 >> 16) | ((tensor_d1 & 0xffffu) << 16));
  g1[3] = (int)((tensor_d1 >> 16) | ((tile_d0 & 0xffffu) << 16));
  g1[4] = (int)(tile_d1 & 0xffffu);                      // tile_dim1, tile_dim2=0
  g1[5] = (int)stride_elems;                             // tensor_dim0_stride
  g1[6] = 0;
  g1[7] = 0;
  int32x4 g2 = {0, 0, 0, 0};
  int32x4 g3 = {0, 0, 0, 0};
  int32x8 g4 = {0, 0, 0, 0, 0, 0, 0, 0};
  __builtin_amdgcn_tensor_load_to_lds(g0, g1, g2, g3, g4, 0);
}
#endif

// ---------------------------------------------------------------------------
// GEMM: C[M,N] (f32) = A[M,K] (bf16, row-major) @ B[N,K]^T (bf16, row-major)
// Block tile 128x128, BK=32, 256 threads = 8 waves (4x2), each wave 32x64
// (2x4 WMMA 16x16x32 bf16 tiles). Tiles staged via TDM (double-buffered)
// when available, else cooperative loads.
// ---------------------------------------------------------------------------
__global__ void __launch_bounds__(256)
gemm_bf16_kernel(const __bf16* __restrict__ A, const __bf16* __restrict__ Bm,
                 float* __restrict__ C, int M, int N, int K) {
  __shared__ __attribute__((aligned(16))) __bf16 sA[2][128][32];
  __shared__ __attribute__((aligned(16))) __bf16 sB[2][128][32];

  const int tid  = threadIdx.x;
  const int wid  = tid >> 5;
  const int lane = tid & 31;
  const int wm   = wid >> 1;      // 0..3  (M direction)
  const int wn   = wid & 1;       // 0..1  (N direction)
  const int h    = lane >> 4;     // half-wave
  const int l16  = lane & 15;

  const int rowC = blockIdx.y * 128;
  const int colC = blockIdx.x * 128;
  const int steps = K / 32;

  v8f acc[2][4] = {};

#if USE_TDM
  if (wid == 0) {
    tdm_load_tile_2d((unsigned)(size_t)&sA[0][0][0],
                     A + (size_t)rowC * K, K, M, K, 32, 128);
    tdm_load_tile_2d((unsigned)(size_t)&sB[0][0][0],
                     Bm + (size_t)colC * K, K, N, K, 32, 128);
  }
#endif

  for (int i = 0; i < steps; ++i) {
    const int cur = i & 1;
#if USE_TDM
    const int nxt = cur ^ 1;
    __syncthreads();                 // everyone done reading buf[nxt] (iter i-1)
    if (wid == 0) {
      if (i + 1 < steps) {
        int k0n = (i + 1) * 32;
        tdm_load_tile_2d((unsigned)(size_t)&sA[nxt][0][0],
                         A + (size_t)rowC * K + k0n, K, M, K, 32, 128);
        tdm_load_tile_2d((unsigned)(size_t)&sB[nxt][0][0],
                         Bm + (size_t)colC * K + k0n, K, N, K, 32, 128);
        __builtin_amdgcn_s_wait_tensorcnt(2);  // step-i loads done (in-order)
      } else {
        __builtin_amdgcn_s_wait_tensorcnt(0);
      }
    }
    __syncthreads();                 // buf[cur] ready for all waves
#else
    const int k0 = i * 32;
    __syncthreads();
    {
      int r  = tid >> 1;
      int ch = (tid & 1) * 16;
      const uint4* ga = (const uint4*)(A + (size_t)(rowC + r) * K + k0 + ch);
      ((uint4*)&sA[cur][r][ch])[0] = ga[0];
      ((uint4*)&sA[cur][r][ch])[1] = ga[1];
      const uint4* gb = (const uint4*)(Bm + (size_t)(colC + r) * K + k0 + ch);
      ((uint4*)&sB[cur][r][ch])[0] = gb[0];
      ((uint4*)&sB[cur][r][ch])[1] = gb[1];
    }
    __syncthreads();
#endif

    // Fragments per ISA layout.
    v16bf af[2], bf[4];
#pragma unroll
    for (int ti = 0; ti < 2; ++ti) {
      int ar = wm * 32 + ti * 16 + l16;
      ((uint4*)&af[ti])[0] = *(const uint4*)&sA[cur][ar][h * 8];
      ((uint4*)&af[ti])[1] = *(const uint4*)&sA[cur][ar][16 + h * 8];
    }
#pragma unroll
    for (int tj = 0; tj < 4; ++tj) {
      int br = wn * 64 + tj * 16 + l16;
      ((uint4*)&bf[tj])[0] = *(const uint4*)&sB[cur][br][h * 8];
      ((uint4*)&bf[tj])[1] = *(const uint4*)&sB[cur][br][16 + h * 8];
    }
#pragma unroll
    for (int ti = 0; ti < 2; ++ti)
#pragma unroll
      for (int tj = 0; tj < 4; ++tj)
        acc[ti][tj] = __builtin_amdgcn_wmma_f32_16x16x32_bf16(
            false, af[ti], false, bf[tj], (short)0, acc[ti][tj], false, false);
  }

  // Epilogue: C-layout VGPR r -> row r + 8*h, col = l16.
#pragma unroll
  for (int ti = 0; ti < 2; ++ti)
#pragma unroll
    for (int tj = 0; tj < 4; ++tj) {
      int colg = colC + wn * 64 + tj * 16 + l16;
#pragma unroll
      for (int r = 0; r < 8; ++r) {
        int rowg = rowC + wm * 32 + ti * 16 + r + 8 * h;
        C[(size_t)rowg * N + colg] = acc[ti][tj][r];
      }
    }
}

// ---------------------------------------------------------------------------
// Partial RoPE (first 16 dims of each head) + f32 -> bf16.
// ---------------------------------------------------------------------------
__global__ void rope_cvt_kernel(const float* __restrict__ X,
                                const int* __restrict__ pos_ids,
                                __bf16* __restrict__ Y, int nh, int total) {
  int idx = blockIdx.x * blockDim.x + threadIdx.x;
  if (idx >= total) return;
  int head = idx % nh;
  int row  = idx / nh;
  float pos = (float)pos_ids[row];
  const float* x = X + (size_t)row * (nh * HD_) + head * HD_;
  __bf16*      y = Y + (size_t)row * (nh * HD_) + head * HD_;
#pragma unroll
  for (int i = 0; i < ROT_ / 2; ++i) {
    float freq = __powf(10000.0f, -(float)(2 * i) / (float)ROT_);
    float ang  = pos * freq;
    float c = cosf(ang), s = sinf(ang);
    float x1 = x[i], x2 = x[i + ROT_ / 2];
    y[i]            = (__bf16)(x1 * c - x2 * s);
    y[i + ROT_ / 2] = (__bf16)(x2 * c + x1 * s);
  }
#pragma unroll
  for (int d = ROT_; d < HD_; ++d) y[d] = (__bf16)x[d];
}

// ---------------------------------------------------------------------------
// V: f32 [b,s,h,d] -> bf16 transposed [b,h,d,s].
// ---------------------------------------------------------------------------
__global__ void vcvt_t_kernel(const float* __restrict__ Vf,
                              __bf16* __restrict__ Vt, int total) {
  int idx = blockIdx.x * blockDim.x + threadIdx.x;
  if (idx >= total) return;
  int d   = idx % HD_;
  int t   = idx / HD_;
  int hh  = t % NKV_;
  int row = t / NKV_;                 // b*S + s
  int s   = row % S_;
  int b   = row / S_;
  Vt[(((size_t)b * NKV_ + hh) * HD_ + d) * S_ + s] = (__bf16)Vf[idx];
}

// ---------------------------------------------------------------------------
// Flash attention (causal), one wave per (b, q-head, 16-row Q block).
// ---------------------------------------------------------------------------
__global__ void __launch_bounds__(128)
attn_kernel(const __bf16* __restrict__ Qb, const __bf16* __restrict__ Kb,
            const __bf16* __restrict__ Vt, __bf16* __restrict__ Ob) {
  __shared__ __attribute__((aligned(16))) __bf16 P[4][16][32];

  const int tid  = threadIdx.x;
  const int wid  = tid >> 5;
  const int lane = tid & 31;
  const int h    = lane >> 4;
  const int l16  = lane & 15;

  const int QB   = S_ / 16;                 // 128 q-blocks per head
  int w    = blockIdx.x * 4 + wid;
  int qblk = w % QB;
  int tmp  = w / QB;
  int qh   = tmp % NH_;
  int b    = tmp / NH_;
  int kvh  = qh / G_;
  int q0   = qblk * 16;

  // Q A-fragments for the two K=32 chunks of HD=64.
  v16bf qf[2];
  {
    const __bf16* qrow = Qb + ((size_t)(b * S_ + q0 + l16) * NH_ + qh) * HD_;
#pragma unroll
    for (int c = 0; c < 2; ++c) {
      ((uint4*)&qf[c])[0] = *(const uint4*)(qrow + c * 32 + h * 8);
      ((uint4*)&qf[c])[1] = *(const uint4*)(qrow + c * 32 + 16 + h * 8);
    }
  }

  v8f o[4] = {};
  float mrow[8], lrow[8];
#pragma unroll
  for (int r = 0; r < 8; ++r) { mrow[r] = -1e38f; lrow[r] = 0.f; }

  for (int j0 = 0; j0 < q0 + 16; j0 += 32) {
    // Prefetch next K/V tiles (lowers to global_prefetch_b8).
    if (j0 + 32 < q0 + 16) {
      __builtin_prefetch(
          Kb + ((size_t)(b * S_ + j0 + 32 + l16) * NKV_ + kvh) * HD_, 0, 1);
      __builtin_prefetch(
          Vt + (((size_t)b * NKV_ + kvh) * HD_ + l16) * S_ + j0 + 32, 0, 1);
    }
    v8f st[2];
#pragma unroll
    for (int sub = 0; sub < 2; ++sub) {
      int key = j0 + sub * 16 + l16;        // B-fragment column = key index
      const __bf16* krow = Kb + ((size_t)(b * S_ + key) * NKV_ + kvh) * HD_;
      v8f sa = {};
#pragma unroll
      for (int c = 0; c < 2; ++c) {
        v16bf kf;
        ((uint4*)&kf)[0] = *(const uint4*)(krow + c * 32 + h * 8);
        ((uint4*)&kf)[1] = *(const uint4*)(krow + c * 32 + 16 + h * 8);
        sa = __builtin_amdgcn_wmma_f32_16x16x32_bf16(
            false, qf[c], false, kf, (short)0, sa, false, false);
      }
#pragma unroll
      for (int r = 0; r < 8; ++r) {
        int rowg = q0 + r + 8 * h;
        int colg = j0 + sub * 16 + l16;
        float v  = sa[r] * 0.125f;          // 1/sqrt(64)
        st[sub][r] = (colg <= rowg) ? v : -1e30f;
      }
    }

    // Online softmax: reductions across the 16-lane half holding each row.
    float alpha[8];
#pragma unroll
    for (int r = 0; r < 8; ++r) {
      float cm = fmaxf(st[0][r], st[1][r]);
#pragma unroll
      for (int d = 1; d < 16; d <<= 1) cm = fmaxf(cm, __shfl_xor(cm, d, 32));
      float mn = fmaxf(mrow[r], cm);
      alpha[r] = __expf(mrow[r] - mn);
      mrow[r]  = mn;
      float p0 = __expf(st[0][r] - mn);
      float p1 = __expf(st[1][r] - mn);
      st[0][r] = p0; st[1][r] = p1;
      float rs = p0 + p1;
#pragma unroll
      for (int d = 1; d < 16; d <<= 1) rs += __shfl_xor(rs, d, 32);
      lrow[r] = lrow[r] * alpha[r] + rs;
    }
#pragma unroll
    for (int t = 0; t < 4; ++t)
#pragma unroll
      for (int r = 0; r < 8; ++r) o[t][r] *= alpha[r];

    // Probs: C-layout -> LDS (row-major 16x32 bf16) -> A-fragment.
#pragma unroll
    for (int r = 0; r < 8; ++r) {
      P[wid][r + 8 * h][l16]      = (__bf16)st[0][r];
      P[wid][r + 8 * h][16 + l16] = (__bf16)st[1][r];
    }
    v16bf pf;
    ((uint4*)&pf)[0] = *(const uint4*)&P[wid][l16][h * 8];
    ((uint4*)&pf)[1] = *(const uint4*)&P[wid][l16][16 + h * 8];

    // PV: B = V chunk (32 keys x 64 d), column-contiguous via Vt.
#pragma unroll
    for (int t = 0; t < 4; ++t) {
      const __bf16* vcol =
          Vt + (((size_t)b * NKV_ + kvh) * HD_ + t * 16 + l16) * S_ + j0 + h * 8;
      v16bf vf;
      ((uint4*)&vf)[0] = ((const uint4*)vcol)[0];
      ((uint4*)&vf)[1] = ((const uint4*)vcol)[2];
      o[t] = __builtin_amdgcn_wmma_f32_16x16x32_bf16(
          false, pf, false, vf, (short)0, o[t], false, false);
    }
  }

  // Finalize and store bf16 in [b,s,qh,d] layout (rows of the O-proj GEMM).
#pragma unroll
  for (int t = 0; t < 4; ++t)
#pragma unroll
    for (int r = 0; r < 8; ++r) {
      int rowg = q0 + r + 8 * h;
      float val = o[t][r] / lrow[r];
      Ob[((size_t)(b * S_ + rowg) * NH_ + qh) * HD_ + t * 16 + l16] = (__bf16)val;
    }
}

// ---------------------------------------------------------------------------
// Launch
// ---------------------------------------------------------------------------
extern "C" void kernel_launch(void* const* d_in, const int* in_sizes, int n_in,
                              void* d_out, int out_size, void* d_ws, size_t ws_size,
                              hipStream_t stream) {
  const float* hs  = (const float*)d_in[0];
  // d_in[1] attention_mask: pure causal -> applied analytically in-kernel.
  const int*   pos = (const int*)d_in[2];
  const float* Wq  = (const float*)d_in[3];
  const float* Wk  = (const float*)d_in[4];
  const float* Wv  = (const float*)d_in[5];
  const float* Wo  = (const float*)d_in[6];
  float* out = (float*)d_out;

  char* ws = (char*)d_ws;
  size_t off = 0;
  auto alloc = [&](size_t bytes) -> void* {
    void* p = ws + off;
    off += (bytes + 255) & ~(size_t)255;
    return p;
  };

  __bf16* hsb   = (__bf16*)alloc((size_t)MROWS * H_ * 2);
  __bf16* wqb   = (__bf16*)alloc((size_t)H_ * H_ * 2);
  __bf16* wkb   = (__bf16*)alloc((size_t)(NKV_ * HD_) * H_ * 2);
  __bf16* wvb   = (__bf16*)alloc((size_t)(NKV_ * HD_) * H_ * 2);
  __bf16* wob   = (__bf16*)alloc((size_t)H_ * H_ * 2);
  float*  qf32  = (float*)alloc((size_t)MROWS * H_ * 4);
  float*  kf32  = (float*)alloc((size_t)MROWS * NKV_ * HD_ * 4);
  float*  vf32  = (float*)alloc((size_t)MROWS * NKV_ * HD_ * 4);
  __bf16* qb    = (__bf16*)alloc((size_t)MROWS * H_ * 2);
  __bf16* kb    = (__bf16*)alloc((size_t)MROWS * NKV_ * HD_ * 2);
  __bf16* vtb   = (__bf16*)alloc((size_t)MROWS * NKV_ * HD_ * 2);
  __bf16* attnb = (__bf16*)alloc((size_t)MROWS * H_ * 2);

  auto cvt = [&](const float* s, __bf16* d, int n) {
    cvt_bf16_kernel<<<(n + 255) / 256, 256, 0, stream>>>(s, d, n);
  };
  cvt(hs, hsb, MROWS * H_);
  cvt(Wq, wqb, H_ * H_);
  cvt(Wk, wkb, NKV_ * HD_ * H_);
  cvt(Wv, wvb, NKV_ * HD_ * H_);
  cvt(Wo, wob, H_ * H_);

  // Projections: C = X @ W^T
  gemm_bf16_kernel<<<dim3(H_ / 128, MROWS / 128), 256, 0, stream>>>(
      hsb, wqb, qf32, MROWS, H_, H_);
  gemm_bf16_kernel<<<dim3((NKV_ * HD_) / 128, MROWS / 128), 256, 0, stream>>>(
      hsb, wkb, kf32, MROWS, NKV_ * HD_, H_);
  gemm_bf16_kernel<<<dim3((NKV_ * HD_) / 128, MROWS / 128), 256, 0, stream>>>(
      hsb, wvb, vf32, MROWS, NKV_ * HD_, H_);

  // RoPE + bf16 repack
  {
    int nq = MROWS * NH_;
    rope_cvt_kernel<<<(nq + 255) / 256, 256, 0, stream>>>(qf32, pos, qb, NH_, nq);
    int nk = MROWS * NKV_;
    rope_cvt_kernel<<<(nk + 255) / 256, 256, 0, stream>>>(kf32, pos, kb, NKV_, nk);
    int nv = MROWS * NKV_ * HD_;
    vcvt_t_kernel<<<(nv + 255) / 256, 256, 0, stream>>>(vf32, vtb, nv);
  }

  // Flash attention: B*NH*(S/16) waves, 4 waves per block.
  {
    int waves = B_ * NH_ * (S_ / 16);
    attn_kernel<<<waves / 4, 128, 0, stream>>>(qb, kb, vtb, attnb);
  }

  // Output projection
  gemm_bf16_kernel<<<dim3(H_ / 128, MROWS / 128), 256, 0, stream>>>(
      attnb, wob, out, MROWS, H_, H_);
}
